// MultiHeadAttention_17884243821067
// MI455X (gfx1250) — matrix operand order repaired and checked by reading, not compiled, not run on hardware
//
#include <hip/hip_runtime.h>
#include <hip/hip_bf16.h>

// ---------------------------------------------------------------------------
// MHA forward for gfx1250 (MI455X).  N=2, L=1024, E=2048, H=16, D=128.
// GEMMs: V_WMMA_F32_16X16X32_BF16 (fp32 accumulate).
// Flash-attention K/V tiles streamed into LDS with the Tensor Data Mover
// (TENSOR_LOAD_TO_LDS, double-buffered, S_WAIT_TENSORCNT pipelined).
// ---------------------------------------------------------------------------

typedef __attribute__((ext_vector_type(8)))  float  v8f;
typedef __attribute__((ext_vector_type(16))) __bf16 v16bf;
typedef __attribute__((ext_vector_type(8)))  __bf16 v8bf;
typedef __attribute__((ext_vector_type(4)))  unsigned int v4u;
typedef __attribute__((ext_vector_type(8)))  int v8i;
typedef __attribute__((ext_vector_type(4)))  int v4i;

union BF16x16 { v16bf v; v8bf h[2]; };

static __device__ inline v8f zero_v8f() {
  v8f z;
#pragma unroll
  for (int i = 0; i < 8; ++i) z[i] = 0.0f;
  return z;
}

static __device__ inline v8f wmma_bf16(v16bf a, v16bf b, v8f c) {
  // (neg_a, A, neg_b, B, c_mod, C, reuse_a, reuse_b)
  return __builtin_amdgcn_wmma_f32_16x16x32_bf16(false, a, false, b,
                                                 (short)0, c, false, false);
}

// A-matrix (16x32 bf16) fragment loader.  ISA layout: lane = m + 16*s holds
// K = 8s+[0..7] (v0..3) and 16+8s+[0..7] (v4..7) of row m -> two 16B loads.
static __device__ inline v16bf load_a_frag(const __bf16* __restrict__ base,
                                           int row, int ldm, int k0, int s) {
  BF16x16 f;
  const __bf16* p = base + (size_t)row * ldm + k0 + s * 8;
  f.h[0] = *(const v8bf*)(p);
  f.h[1] = *(const v8bf*)(p + 16);
  return f.v;
}

// B-matrix (32x16 bf16) fragment loader.  ISA layout: lane = c + 16*s holds
// K = 8s+[0..7] (v0..3) and 16+8s+[0..7] (v4..7) of column c -> two 16B loads.
// `row` is the source row holding column c's data (B stored as B^T row-major).
static __device__ inline v16bf load_b_frag(const __bf16* __restrict__ base,
                                           int row, int ldm, int k0, int s) {
  BF16x16 f;
  const __bf16* p = base + (size_t)row * ldm + k0 + s * 8;
  f.h[0] = *(const v8bf*)(p);
  f.h[1] = *(const v8bf*)(p + 16);
  return f.v;
}

// LDS aperture places the wave-relative LDS byte offset in the low 32 bits of
// a generic address (ISA 10.2).
static __device__ inline unsigned lds_byte_off(const void* p) {
  return (unsigned)(uintptr_t)p;
}

// Tensor Data Mover: 2D tile load, 2-byte elements.
//   tile_w elements per row, tile_h rows, source row stride in elements.
// D# bit packing per CDNA5 ISA 8.3/8.4 (group0 128b, group1 256b).
static __device__ inline void tdm_load_2d_bf16(unsigned lds_addr,
                                               const void* gptr,
                                               unsigned tile_w, unsigned tile_h,
                                               unsigned stride_elems) {
  unsigned long long ga = (unsigned long long)(uintptr_t)gptr;
  v4u g0;
  g0[0] = 1u;                                   // count=1 (valid user D#)
  g0[1] = lds_addr;                             // lds_addr[31:0]
  g0[2] = (unsigned)(ga & 0xffffffffu);         // global_addr[31:0]
  g0[3] = (unsigned)((ga >> 32) & 0x01ffffffu)  // global_addr[56:32]
          | (2u << 30);                         // type=2 ("image")
  v8i g1;
  g1[0] = (int)(1u << 16);                      // data_size=1 -> 2 bytes
  g1[1] = (int)((tile_w & 0xffffu) << 16);      // tensor_dim0[15:0] @ [63:48]
  g1[2] = (int)(((tile_w >> 16) & 0xffffu)      // tensor_dim0[31:16]
          | ((tile_h & 0xffffu) << 16));        // tensor_dim1[15:0]
  g1[3] = (int)(((tile_h >> 16) & 0xffffu)      // tensor_dim1[31:16]
          | ((tile_w & 0xffffu) << 16));        // tile_dim0 @ [127:112]
  g1[4] = (int)(tile_h & 0xffffu);              // tile_dim1 (tile_dim2=0)
  g1[5] = (int)stride_elems;                    // tensor_dim0_stride[31:0]
  g1[6] = 0;                                    // stride[47:32], dim1_stride lo
  g1[7] = 0;
  v4i z4;
  z4[0] = 0; z4[1] = 0; z4[2] = 0; z4[3] = 0;
  v8i z8;
#pragma unroll
  for (int i = 0; i < 8; ++i) z8[i] = 0;
  // 6-arg form (clang-23 / amdgpu-toolchain): (g0, g1, g2, g3, g4, cpol)
  __builtin_amdgcn_tensor_load_to_lds(g0, g1, z4, z4, z8, 0);
}

// ---------------------------------------------------------------------------
// Kernel 1: convert weights fp32 -> bf16
// ---------------------------------------------------------------------------
__global__ void __launch_bounds__(256)
cvt_weights_kernel(const float* __restrict__ Wq, const float* __restrict__ Wk,
                   const float* __restrict__ Wv, const float* __restrict__ Wo,
                   __bf16* __restrict__ WqB, __bf16* __restrict__ WkB,
                   __bf16* __restrict__ WvB, __bf16* __restrict__ WoB) {
  int i = blockIdx.x * 256 + threadIdx.x;
  if (i < 2048 * 2048) WoB[i] = (__bf16)Wo[i];
  if (i < 128 * 128) {
    WqB[i] = (__bf16)Wq[i];
    WkB[i] = (__bf16)Wk[i];
    WvB[i] = (__bf16)Wv[i];
  }
}

// ---------------------------------------------------------------------------
// Kernel 2: fused RoPE + per-head projection.
//   X: [N, L, H, D] fp32.   Wb: [128,128] bf16 (row o, col d).
//   out[n,l,h,o] = sum_d rope(x)[n,l,h,d] * W[o,d]
//   store_trans=0: Out[(n*16+h)*1024 + l][o]   ([NH][L][D], for Q/K)
//   store_trans=1: Out[(n*16+h)*128 + o][l]    ([NH][D][L], for V)
//   RoPE per reference: r[d] = x[d]*cos(t*f[d&63]) + x[127-d]*sin(t*f[d&63]),
//   f[j] = 10000^(-j/64) = exp(-j*ln(1e4)/64).
// One wave per (n,l): 16 rows = 16 heads, shared t = l.
// ---------------------------------------------------------------------------
__global__ void __launch_bounds__(256)
proj_rope_kernel(const float* __restrict__ X, const __bf16* __restrict__ Wb,
                 __bf16* __restrict__ Out, int do_rope, int store_trans) {
  const int wave = (blockIdx.x * 256 + threadIdx.x) >> 5;  // 0..2047
  const int lane = threadIdx.x & 31;
  const int n = wave >> 10;
  const int l = wave & 1023;
  const int m = lane & 15;   // head / B column
  const int s = lane >> 4;
  const float t = (float)l;

  const float* xrow = X + (((size_t)(n * 1024 + l)) * 16 + m) * 128;

  v16bf a[4];
#pragma unroll
  for (int kb = 0; kb < 4; ++kb) {
    BF16x16 f;
#pragma unroll
    for (int half = 0; half < 2; ++half) {
      const int d0 = kb * 32 + half * 16 + s * 8;
#pragma unroll
      for (int j = 0; j < 8; ++j) {
        const int d = d0 + j;
        float x = xrow[d];
        float r = x;
        if (do_rope) {
          float xf = xrow[127 - d];
          float fr = __expf(-0.14391156831212787f * (float)(d & 63));
          float ang = t * fr;
          r = x * cosf(ang) + xf * sinf(ang);
        }
        f.v[half * 8 + j] = (__bf16)r;
      }
    }
    a[kb] = f.v;
  }

#pragma unroll
  for (int nt = 0; nt < 8; ++nt) {
    v8f acc = zero_v8f();
#pragma unroll
    for (int kb = 0; kb < 4; ++kb) {
      v16bf b = load_b_frag(Wb, nt * 16 + m, 128, kb * 32, s);
      acc = wmma_bf16(a[kb], b, acc);
    }
#pragma unroll
    for (int r = 0; r < 8; ++r) {
      const int head = r + 8 * s;  // C/D layout: lane half selects row group
      const int col = nt * 16 + m;
      size_t idx =
          store_trans
              ? (((size_t)(n * 16 + head) * 128 + col) * 1024 + (size_t)l)
              : (((size_t)(n * 16 + head) * 1024 + l) * 128 + (size_t)col);
      Out[idx] = (__bf16)acc[r];
    }
  }
}

// ---------------------------------------------------------------------------
// Kernel 3: flash attention per (n,h), one wave per 16-query tile.
//   Qp,Kp: [NH][L][128] bf16.  VpT: [NH][128][L] bf16.  Mask: [N][L][L] i32.
//   AO: [N*L][2048] bf16  (col = h*128 + d), input to output projection.
// K tile (32x128, contiguous 8KB) and V tile (128x32, stride-1024 2D) are
// streamed into LDS by the TDM, double-buffered, S_WAIT_TENSORCNT pipelined.
// ---------------------------------------------------------------------------
__global__ void __launch_bounds__(32)
flash_attn_kernel(const __bf16* __restrict__ Qp, const __bf16* __restrict__ Kp,
                  const __bf16* __restrict__ VpT, const int* __restrict__ Mask,
                  __bf16* __restrict__ AO) {
  const int nh = blockIdx.x;  // 0..31
  const int n = nh >> 4;
  const int h = nh & 15;
  const int lane = threadIdx.x & 31;
  const int qb = blockIdx.y * 16;  // query tile base (blockIdx.y = 0..63)
  const int m = lane & 15;
  const int s = lane >> 4;
  const float scale = 0.02209708691207961f;  // 1/sqrt(2048)

  const __bf16* q = Qp + (size_t)nh * 1024 * 128;
  const __bf16* kp = Kp + (size_t)nh * 1024 * 128;
  const __bf16* vt = VpT + (size_t)nh * 128 * 1024;
  const int* mk = Mask + (size_t)n * 1024 * 1024;

  // Per-wave LDS: double-buffered K (32x128) and V (128x32) tiles + P bounce.
  __shared__ __align__(16) __bf16 kbuf[2][32 * 128];  // 2 x 8KB
  __shared__ __align__(16) __bf16 vbuf[2][128 * 32];  // 2 x 8KB
  __shared__ __align__(16) __bf16 pbuf[16 * 32];      // 1KB

  v16bf aq[4];
#pragma unroll
  for (int kb = 0; kb < 4; ++kb) aq[kb] = load_a_frag(q, qb + m, 128, kb * 32, s);

  v8f o[8];
  float rm[8], rl[8];
#pragma unroll
  for (int i = 0; i < 8; ++i) {
    o[i] = zero_v8f();
    rm[i] = -3.0e38f;
    rl[i] = 0.0f;
  }

  // Prologue: TDM-load tile 0 (K then V -> TENSORcnt = 2).
  tdm_load_2d_bf16(lds_byte_off(&kbuf[0][0]), kp, 128, 32, 128);
  tdm_load_2d_bf16(lds_byte_off(&vbuf[0][0]), vt, 32, 128, 1024);

  for (int kb2 = 0; kb2 < 32; ++kb2) {
    const int kbase = kb2 * 32;
    if (kb2 + 1 < 32) {
      const int nb = (kb2 + 1) & 1;
      const int nkb = kbase + 32;
      // Prefetch next K/V tiles; then wait until only those 2 are in flight
      // (TENSORcnt ops retire in order -> current tile has landed).
      tdm_load_2d_bf16(lds_byte_off(&kbuf[nb][0]), kp + (size_t)nkb * 128,
                       128, 32, 128);
      tdm_load_2d_bf16(lds_byte_off(&vbuf[nb][0]), vt + nkb, 32, 128, 1024);
      __builtin_amdgcn_s_wait_tensorcnt(2);
    } else {
      __builtin_amdgcn_s_wait_tensorcnt(0);
    }
    asm volatile("" ::: "memory");
    const __bf16* kt = &kbuf[kb2 & 1][0];     // [32 keys][128 d]
    const __bf16* vtile = &vbuf[kb2 & 1][0];  // [128 d][32 keys]

    // S = Q . K^T : two 16x16 tiles over 32 keys, inner K(dim)=128.
    v8f s0 = zero_v8f(), s1 = zero_v8f();
#pragma unroll
    for (int kb = 0; kb < 4; ++kb) {
      v16bf b = load_b_frag(kt, m, 128, kb * 32, s);
      s0 = wmma_bf16(aq[kb], b, s0);
    }
#pragma unroll
    for (int kb = 0; kb < 4; ++kb) {
      v16bf b = load_b_frag(kt, 16 + m, 128, kb * 32, s);
      s1 = wmma_bf16(aq[kb], b, s1);
    }
    // Mask + online softmax.  Row r+8s of the tile lives in this lane's half.
#pragma unroll
    for (int r = 0; r < 8; ++r) {
      const int qrow = qb + r + 8 * s;
      const size_t mrow = (size_t)qrow * 1024 + kbase + m;
      float e0 = (mk[mrow] != 0) ? s0[r] : -1.0e20f;
      float e1 = (mk[mrow + 16] != 0) ? s1[r] : -1.0e20f;
      e0 *= scale;
      e1 *= scale;
      float bm = fmaxf(e0, e1);
#pragma unroll
      for (int off = 8; off >= 1; off >>= 1)
        bm = fmaxf(bm, __shfl_xor(bm, off, 16));
      const float mnew = fmaxf(rm[r], bm);
      const float alpha = __expf(rm[r] - mnew);
      rm[r] = mnew;
      const float p0 = __expf(e0 - mnew);
      const float p1 = __expf(e1 - mnew);
      float bs = p0 + p1;
#pragma unroll
      for (int off = 8; off >= 1; off >>= 1) bs += __shfl_xor(bs, off, 16);
      rl[r] = rl[r] * alpha + bs;
#pragma unroll
      for (int nt = 0; nt < 8; ++nt) o[nt][r] *= alpha;
      const int prow = r + 8 * s;
      pbuf[prow * 32 + m] = (__bf16)p0;
      pbuf[prow * 32 + 16 + m] = (__bf16)p1;
    }
    // DS ops are in-order per wave; fence to keep the compiler honest.
    asm volatile("s_wait_dscnt 0x0" ::: "memory");
    // Re-read P in A-fragment layout (16 queries x 32 keys).
    BF16x16 pf;
    pf.h[0] = *(const v8bf*)(pbuf + m * 32 + s * 8);
    pf.h[1] = *(const v8bf*)(pbuf + m * 32 + 16 + s * 8);
    // O += P . V  (V tile in LDS: row = d, 32 keys per row).
#pragma unroll
    for (int nt = 0; nt < 8; ++nt) {
      v16bf bv = load_b_frag(vtile, nt * 16 + m, 32, 0, s);
      o[nt] = wmma_bf16(pf.v, bv, o[nt]);
    }
  }

#pragma unroll
  for (int r = 0; r < 8; ++r) {
    const float inv = 1.0f / rl[r];
    const int qrow = qb + r + 8 * s;
#pragma unroll
    for (int nt = 0; nt < 8; ++nt) {
      const int col = h * 128 + nt * 16 + m;
      AO[((size_t)(n * 1024 + qrow)) * 2048 + col] = (__bf16)(o[nt][r] * inv);
    }
  }
}

// ---------------------------------------------------------------------------
// Kernel 4: output projection.  Out[row, o] = sum_k AO[row,k]*Wo[o,k] + bo[o]
//   A: [2048][2048] bf16.  WoB: [2048][2048] bf16.  Out: fp32.
// One wave computes a 16(M) x 128(N) tile, K loop = 64 x 32.
// ---------------------------------------------------------------------------
__global__ void __launch_bounds__(256)
out_proj_kernel(const __bf16* __restrict__ A, const __bf16* __restrict__ WoB,
                const float* __restrict__ bo, float* __restrict__ Out) {
  const int w = (blockIdx.x * 256 + threadIdx.x) >> 5;  // 0..2047
  const int lane = threadIdx.x & 31;
  const int mt = w >> 4;  // 0..127
  const int ng = w & 15;  // 0..15 column group of 128
  const int m = lane & 15;
  const int s = lane >> 4;

  v8f acc[8];
#pragma unroll
  for (int i = 0; i < 8; ++i) acc[i] = zero_v8f();

  for (int kb = 0; kb < 64; ++kb) {
    v16bf a = load_a_frag(A, mt * 16 + m, 2048, kb * 32, s);
#pragma unroll
    for (int nt = 0; nt < 8; ++nt) {
      v16bf b = load_b_frag(WoB, ng * 128 + nt * 16 + m, 2048, kb * 32, s);
      acc[nt] = wmma_bf16(a, b, acc[nt]);
    }
  }
#pragma unroll
  for (int nt = 0; nt < 8; ++nt) {
#pragma unroll
    for (int r = 0; r < 8; ++r) {
      const int row = mt * 16 + r + 8 * s;
      const int col = ng * 128 + nt * 16 + m;
      Out[(size_t)row * 2048 + col] = acc[nt][r] + bo[col];
    }
  }
}

// ---------------------------------------------------------------------------
// Host side
// ---------------------------------------------------------------------------
extern "C" void kernel_launch(void* const* d_in, const int* in_sizes, int n_in,
                              void* d_out, int out_size, void* d_ws,
                              size_t ws_size, hipStream_t stream) {
  const float* values = (const float*)d_in[0];
  const float* keys = (const float*)d_in[1];
  const float* queries = (const float*)d_in[2];
  const int* mask = (const int*)d_in[3];
  const float* Wv = (const float*)d_in[4];
  const float* Wk = (const float*)d_in[5];
  const float* Wq = (const float*)d_in[6];
  const float* Wo = (const float*)d_in[7];
  const float* bo = (const float*)d_in[8];
  float* out = (float*)d_out;

  // Workspace layout (bf16 elements).  Total ~40.1 MiB.
  const size_t SZ = 32ull * 1024 * 128;  // 4,194,304 elems per tensor
  __bf16* qp = (__bf16*)d_ws;            // [NH][L][D]
  __bf16* kp = qp + SZ;                  // [NH][L][D]
  __bf16* vpt = kp + SZ;                 // [NH][D][L]
  __bf16* ao = vpt + SZ;                 // [N*L][E]
  __bf16* wob = ao + SZ;                 // [E][E]
  __bf16* wqb = wob + 2048ull * 2048;    // [D][D]
  __bf16* wkb = wqb + 128 * 128;
  __bf16* wvb = wkb + 128 * 128;

  cvt_weights_kernel<<<dim3((2048 * 2048 + 255) / 256), dim3(256), 0, stream>>>(
      Wq, Wk, Wv, Wo, wqb, wkb, wvb, wob);

  // 2048 waves (one per (n,l)), 8 waves per 256-thread block.
  proj_rope_kernel<<<dim3(256), dim3(256), 0, stream>>>(queries, wqb, qp, 1, 0);
  proj_rope_kernel<<<dim3(256), dim3(256), 0, stream>>>(keys, wkb, kp, 1, 0);
  proj_rope_kernel<<<dim3(256), dim3(256), 0, stream>>>(values, wvb, vpt, 0, 1);

  // 32 (n,h) x 64 query tiles; 1 wave per block (per-wave TDM double buffer).
  flash_attn_kernel<<<dim3(32, 64), dim3(32), 0, stream>>>(qp, kp, vpt, mask,
                                                           ao);

  // 2048 waves: 128 M-tiles x 16 N-groups; 8 waves per block.
  out_proj_kernel<<<dim3(256), dim3(256), 0, stream>>>(ao, wob, bo, out);
}
